// GetAdaptivePseudoMaskSLFCAMS_27530740367900
// MI455X (gfx1250) — matrix operand order repaired and checked by reading, not compiled, not run on hardware
//
#include <hip/hip_runtime.h>
#include <hip/hip_bf16.h>

// ---------------------------------------------------------------------------
// GetAdaptivePseudoMask for MI455X (gfx1250, wave32, WGP).
//   grid  = one workgroup per image (B=256)
//   block = 256 threads (8 waves)
// HBM traffic: 2x 64MB read (double-buffered async-to-LDS staging) + 64MB
// write -> ~8us floor at 23.3 TB/s. All reductions live in LDS.
// CDNA5 paths: GLOBAL_LOAD_ASYNC_TO_LDS_B128 (+ s_wait_asynccnt) staging,
// v_wmma_f32_16x16x32_f16 for exact 256-bin prefix sums.
// ---------------------------------------------------------------------------

#define HW      65536   // 256x256 pixels per image
#define TILE    1024    // floats per staged tile (4 KB)
#define NTILES  (HW / TILE)
#define IGNORE_IDX (-255)

typedef _Float16 f16;
typedef __attribute__((ext_vector_type(16))) _Float16 v16h;
typedef __attribute__((ext_vector_type(8)))  float    v8f;

// Exact parameter types of __builtin_amdgcn_global_load_async_to_lds_b128
// (from the round-2 diagnostic): non-const v4i in AS(1), v4i in AS(3).
typedef int v4i_vs __attribute__((vector_size(16)));
typedef __attribute__((address_space(1))) v4i_vs gv4i_t;
typedef __attribute__((address_space(3))) v4i_vs lv4i_t;
typedef __attribute__((address_space(3))) void       l3v_t;
typedef __attribute__((address_space(3))) const void l3cv_t;

// LDS byte offset of a __shared__ object (generic -> addrspace(3) -> int).
__device__ __forceinline__ unsigned ldsAddrOf(const void* p) {
  return (unsigned)(unsigned long long)(l3cv_t*)p;
}

// ---- async memory -> LDS copy, 16B per lane (ASYNCcnt-tracked) -------------
#if defined(__has_builtin)
#  if __has_builtin(__builtin_amdgcn_global_load_async_to_lds_b128)
#    define ASYNC_B128_BUILTIN 1
#  endif
#  if __has_builtin(__builtin_amdgcn_s_wait_asynccnt)
#    define WAIT_ASYNC(n) __builtin_amdgcn_s_wait_asynccnt(n)
#  endif
#endif
#ifndef WAIT_ASYNC
#  define WAIT_ASYNC(n) asm volatile("s_wait_asynccnt " #n ::: "memory")
#endif

// Copy 16 bytes: LDS[ldsPtr] = MEM[gbase + voff].
__device__ __forceinline__ void asyncCopyB128(unsigned long long gbase,
                                              unsigned voff, void* ldsPtr) {
#if defined(ASYNC_B128_BUILTIN)
  __builtin_amdgcn_global_load_async_to_lds_b128(
      (gv4i_t*)(gbase + (unsigned long long)voff),
      (lv4i_t*)(l3v_t*)ldsPtr,
      0, 0);
#else
  // VGLOBAL GLOBAL_LOAD_ASYNC_TO_LDS_B128: VDST = LDS byte address VGPR,
  // VADDR = 32-bit offset VGPR, SADDR = 64-bit base SGPR pair (GVS mode).
  asm volatile("global_load_async_to_lds_b128 %0, %1, %2"
               :: "v"(ldsAddrOf(ldsPtr)), "v"(voff), "s"(gbase)
               : "memory");
#endif
}

__device__ __forceinline__ unsigned pcgHash(unsigned v) {
  v = v * 747796405u + 2891336453u;
  unsigned w = ((v >> ((v >> 28u) + 4u)) ^ v) * 277803737u;
  return (w >> 22u) ^ w;
}

// Keep the 10 largest keys, sorted descending, fully unrolled (stays in VGPRs).
__device__ __forceinline__ void bubbleInsert(unsigned (&l)[10], unsigned key) {
#pragma unroll
  for (int s = 0; s < 10; ++s) {
    unsigned a  = l[s];
    unsigned mx = key > a ? key : a;
    unsigned mn = key > a ? a : key;
    l[s] = mx;
    key  = mn;
  }
}

__device__ __forceinline__ void shiftOut(unsigned (&l)[10]) {
#pragma unroll
  for (int s = 0; s < 9; ++s) l[s] = l[s + 1];
  l[9] = 0u;
}

__global__ __launch_bounds__(256)
void adaptive_pseudo_mask_kernel(const float* __restrict__ x, int* __restrict__ out) {
  __shared__ __align__(16) float stage[2][TILE];   // async double buffer (8 KB)
  __shared__ int      histI[256];
  __shared__ float    parts[5][256];               // f16-exact digit planes
  __shared__ float    csum[256], cvsum[256];
  __shared__ float    offsH[16], offsV[16];
  __shared__ float    varB[256];
  __shared__ float    thrLi;
  __shared__ unsigned maskBits[2][2048];           // dilated fg/bg bitsets (16 KB)
  __shared__ unsigned selCell[2];
  __shared__ unsigned short selIdx[2][10];
  __shared__ int      selCnt[2];

  const int tid = threadIdx.x;
  const int img = blockIdx.x;
  const unsigned long long gbase = (unsigned long long)(x) + (unsigned long long)img * (HW * 4ull);
  const unsigned laneByte = (unsigned)tid * 16u;   // this thread's 16B slot in a tile
  const unsigned imgSalt  = (unsigned)img * 0x9E3779B9u;

  // ---- init ----
  histI[tid] = 0;
#pragma unroll
  for (int i = 0; i < 8; ++i) {
    maskBits[0][tid + 256 * i] = 0u;
    maskBits[1][tid + 256 * i] = 0u;
  }
  if (tid == 0) { selCnt[0] = 0; selCnt[1] = 0; }
  __syncthreads();

  // =========================================================================
  // Pass 1: histogram. Each thread async-copies and consumes its own 16B per
  // tile, so the double buffer needs only per-wave s_wait_asynccnt (ASYNC ops
  // retire in order) -- zero workgroup barriers inside the streaming loop.
  // =========================================================================
  asyncCopyB128(gbase, laneByte, &stage[0][tid * 4]);            // tile 0
  for (int t = 0; t < NTILES; ++t) {
    if (t + 1 < NTILES) {
      asyncCopyB128(gbase, (unsigned)(t + 1) * (TILE * 4u) + laneByte,
                    &stage[(t + 1) & 1][tid * 4]);
      WAIT_ASYNC(1);   // tile t complete, tile t+1 still in flight
    } else {
      WAIT_ASYNC(0);
    }
    const float4 f = *reinterpret_cast<const float4*>(&stage[t & 1][tid * 4]);
    float fv[4] = {f.x, f.y, f.z, f.w};
#pragma unroll
    for (int e = 0; e < 4; ++e) {
      int u = (int)floorf(fv[e] * 255.0f);
      u = u < 0 ? 0 : (u > 255 ? 255 : u);
      atomicAdd(&histI[u], 1);
    }
  }
  __syncthreads();

  // =========================================================================
  // 256-bin prefix sums via WMMA (exact f16-digit decomposition):
  //   csum  = cumsum(h),     h = 256*hA + hB            (hA,hB <= 256: f16 exact)
  //   cvsum = cumsum(bin*h), g = 2048^2*g2 + 2048*g1 + g0 (digits <= 2047)
  // Row scans of the 16x16 bin matrix = M x U (upper-triangular 0/1), two
  // digit planes packed into K=32 with the digit weight folded into B.
  // =========================================================================
  {
    unsigned h = (unsigned)histI[tid];
    unsigned g = h * (unsigned)tid;            // <= 255*65536 < 2^24
    parts[0][tid] = (float)(h >> 8);           // hA
    parts[1][tid] = (float)(h & 255u);         // hB
    parts[2][tid] = (float)(g >> 22);          // g2
    parts[3][tid] = (float)((g >> 11) & 2047u);// g1
    parts[4][tid] = (float)(g & 2047u);        // g0
  }
  __syncthreads();

  if (tid < 32) {  // wave 0 only: EXEC all-ones as WMMA requires
    const int half = tid >> 4;
    const int mrow = tid & 15;       // A row
    const int ncol = tid & 15;       // B/D column
    v16h aH, aG01, aG2, bH, bG01, bG2;
#pragma unroll
    for (int j = 0; j < 16; ++j) {
      const int vg = j >> 1, lohi = j & 1;
      // A k order matches B: lanes 0-15 hold K=j, lanes 16-31 hold K=j+16
      const int k = j + 16 * half; (void)vg; (void)lohi;
      const float ah  = (k < 16) ? parts[0][mrow * 16 + k] : parts[1][mrow * 16 + (k - 16)];
      const float a01 = (k < 16) ? parts[3][mrow * 16 + k] : parts[4][mrow * 16 + (k - 16)];
      const float a2  = (k < 16) ? parts[2][mrow * 16 + k] : 0.0f;
      aH[j]   = (f16)ah;
      aG01[j] = (f16)a01;
      aG2[j]  = (f16)a2;
      // B rows: lanes 0-15 hold K=j, lanes 16-31 hold K=j+16 (digit-1 plane).
      const float u = (j <= ncol) ? 1.0f : 0.0f;
      bH[j]   = (f16)((half == 0) ? 256.0f * u : u);   // weight of hA vs hB
      bG01[j] = (f16)((half == 0) ? 2048.0f * u : u);  // weight of g1 vs g0
      bG2[j]  = (f16)((half == 0) ? u : 0.0f);         // g2 alone, K 16..31 zero
    }
    v8f z = {};
    v8f Dh  = __builtin_amdgcn_wmma_f32_16x16x32_f16(false, aH,   false, bH,   (short)0, z, false, false);
    v8f D01 = __builtin_amdgcn_wmma_f32_16x16x32_f16(false, aG01, false, bG01, (short)0, z, false, false);
    v8f D2  = __builtin_amdgcn_wmma_f32_16x16x32_f16(false, aG2,  false, bG2,  (short)0, z, false, false);
#pragma unroll
    for (int v = 0; v < 8; ++v) {      // D layout: M = v + 8*half, N = lane&15
      const int r  = v + 8 * half;
      const int tt = r * 16 + ncol;
      csum[tt]  = Dh[v];
      cvsum[tt] = 4194304.0f * D2[v] + D01[v];  // 2048^2 weight applied in f32 (exact)
    }
  }
  __syncthreads();
  if (tid == 0) {  // 16 row offsets: tiny exclusive scan of row totals
    float aH2 = 0.0f, aV2 = 0.0f;
#pragma unroll
    for (int r = 0; r < 16; ++r) {
      offsH[r] = aH2; offsV[r] = aV2;
      aH2 += csum[r * 16 + 15];
      aV2 += cvsum[r * 16 + 15];
    }
  }
  __syncthreads();
  csum[tid]  += offsH[tid >> 4];
  cvsum[tid] += offsV[tid >> 4];
  __syncthreads();

  // ---- Otsu between-class variance per candidate threshold ----
  if (tid < 255) {
    const float nT = 65536.0f;
    const float w0 = csum[tid];
    const float w1 = nT - w0;
    const float s0 = cvsum[tid];
    const float sT = cvsum[255];
    const float m0 = s0 / fmaxf(w0, 1e-12f);
    const float m1 = (sT - s0) / fmaxf(w1, 1e-12f);
    const float d  = m0 - m1;
    varB[tid] = w0 * w1 * d * d;
  }
  __syncthreads();

  if (tid == 0) {
    int amax = 0; float best = varB[0];
    for (int i = 1; i < 255; ++i) { if (varB[i] > best) { best = varB[i]; amax = i; } }
    float ot = (float)amax;
    ot = fminf(fmaxf(ot, 1.0f), 254.0f);
    int imn = 255;
    for (int i = 0; i < 256; ++i) { if (histI[i] > 0) { imn = i; break; } }
    const float im = (float)imn;
    // Li minimum cross-entropy iteration (shifted domain, tol=0.5, <=64 iters)
    const float nT = 65536.0f;
    const float sT = cvsum[255] - im * nT;
    float tc = ot - im, tp = tc + 10.0f;
    int it = 0;
    while (fabsf(tc - tp) > 0.5f && it < 64) {
      int idx = (int)floorf(tc + im);
      idx = idx < 0 ? 0 : (idx > 255 ? 255 : idx);
      const float nb = csum[idx];
      const float sb = cvsum[idx] - im * nb;
      const float nf = nT - nb, sf = sT - sb;
      const float mb = (nb > 0.0f) ? (sb / fmaxf(nb, 1.0f)) : 0.0f;
      const float mf = sf / fmaxf(nf, 1.0f);
      const float eps = 1e-12f;
      float tn;
      if (mb < eps) tn = mf * 0.5f;
      else tn = (mb - mf) / (logf(fmaxf(mb, eps)) - logf(fmaxf(mf, eps)));
      tp = tc; tc = tn; ++it;
    }
    thrLi = tc + im;
  }
  __syncthreads();

  // =========================================================================
  // Pass 2: per-pixel random keys, local top-10 per thread (fg and bg).
  // key = [31]=valid | [30:16]=hash score | [15:0]=pixel index (unique).
  // Same barrier-free async-staged streaming loop.
  // =========================================================================
  const float li = thrLi;
  unsigned kf[10], kb[10];
#pragma unroll
  for (int s = 0; s < 10; ++s) { kf[s] = 0u; kb[s] = 0u; }

  asyncCopyB128(gbase, laneByte, &stage[0][tid * 4]);            // tile 0
  for (int t = 0; t < NTILES; ++t) {
    if (t + 1 < NTILES) {
      asyncCopyB128(gbase, (unsigned)(t + 1) * (TILE * 4u) + laneByte,
                    &stage[(t + 1) & 1][tid * 4]);
      WAIT_ASYNC(1);
    } else {
      WAIT_ASYNC(0);
    }
    const float4 f = *reinterpret_cast<const float4*>(&stage[t & 1][tid * 4]);
    float fv[4] = {f.x, f.y, f.z, f.w};
#pragma unroll
    for (int e = 0; e < 4; ++e) {
      int u = (int)floorf(fv[e] * 255.0f);
      u = u < 0 ? 0 : (u > 255 ? 255 : u);
      const unsigned p   = (unsigned)(t * TILE + tid * 4 + e);
      const unsigned hsh = pcgHash(p ^ imgSalt);
      const unsigned key = 0x80000000u | (hsh & 0x7FFF0000u) | p;
      if ((float)u > li) bubbleInsert(kf, key);
      else               bubbleInsert(kb, key);
    }
  }

  // ---- global top-10 selection: 10 rounds of ds_max tournaments ----
  for (int r = 0; r < 10; ++r) {
    if (tid == 0) { selCell[0] = 0u; selCell[1] = 0u; }
    __syncthreads();
    const unsigned cF = kf[0], cB = kb[0];
    if (cF) atomicMax(&selCell[0], cF);
    if (cB) atomicMax(&selCell[1], cB);
    __syncthreads();
    const unsigned wF = selCell[0], wB = selCell[1];
    if (tid == 0) {
      if (wF >> 31) selIdx[0][selCnt[0]++] = (unsigned short)(wF & 0xFFFFu);
      if (wB >> 31) selIdx[1][selCnt[1]++] = (unsigned short)(wB & 0xFFFFu);
    }
    if (cF && cF == wF) shiftOut(kf);   // keys unique -> exactly one winner advances
    if (cB && cB == wB) shiftOut(kb);
    __syncthreads();
  }

  // ---- dilate (3x3 SAME) by stamping each selected point's neighborhood ----
  if (tid < selCnt[0]) {
    const unsigned p = selIdx[0][tid];
    const int y = (int)(p >> 8), xx0 = (int)(p & 255u);
#pragma unroll
    for (int dy = -1; dy <= 1; ++dy)
#pragma unroll
      for (int dx = -1; dx <= 1; ++dx) {
        const int yy = y + dy, xx = xx0 + dx;
        if (yy >= 0 && yy < 256 && xx >= 0 && xx < 256) {
          const unsigned q = (unsigned)(yy * 256 + xx);
          atomicOr(&maskBits[0][q >> 5], 1u << (q & 31u));
        }
      }
  }
  {
    const int bt = tid - 64;
    if (bt >= 0 && bt < selCnt[1]) {
      const unsigned p = selIdx[1][bt];
      const int y = (int)(p >> 8), xx0 = (int)(p & 255u);
#pragma unroll
      for (int dy = -1; dy <= 1; ++dy)
#pragma unroll
        for (int dx = -1; dx <= 1; ++dx) {
          const int yy = y + dy, xx = xx0 + dx;
          if (yy >= 0 && yy < 256 && xx >= 0 && xx < 256) {
            const unsigned q = (unsigned)(yy * 256 + xx);
            atomicOr(&maskBits[1][q >> 5], 1u << (q & 31u));
          }
        }
    }
  }
  __syncthreads();

  // ---- emit seeds: overlap -> -255, fg -> 1, bg -> 0, else -255 (B128) ----
  int* __restrict__ gout = out + (size_t)img * HW;
  for (int i = 0; i < 64; ++i) {
    const int p  = i * 1024 + tid * 4;
    const unsigned fw = maskBits[0][p >> 5] >> (p & 31);
    const unsigned bw = maskBits[1][p >> 5] >> (p & 31);
    int4 o;
    int* ov = &o.x;
#pragma unroll
    for (int e = 0; e < 4; ++e) {
      const int fbit = (int)((fw >> e) & 1u);
      const int bbit = (int)((bw >> e) & 1u);
      ov[e] = (fbit && !bbit) ? 1 : ((bbit && !fbit) ? 0 : IGNORE_IDX);
    }
    *reinterpret_cast<int4*>(gout + p) = o;
  }
}

extern "C" void kernel_launch(void* const* d_in, const int* in_sizes, int n_in,
                              void* d_out, int out_size, void* d_ws, size_t ws_size,
                              hipStream_t stream) {
  (void)n_in; (void)out_size; (void)d_ws; (void)ws_size;
  const float* xin = (const float*)d_in[0];
  int* outp = (int*)d_out;          // reference output dtype is int32
  int nimg = in_sizes[0] / HW;
  if (nimg < 1) nimg = 1;
  adaptive_pseudo_mask_kernel<<<dim3(nimg), dim3(256), 0, stream>>>(xin, outp);
}
